// Model_11209864642616
// MI455X (gfx1250) — matrix operand order repaired and checked by reading, not compiled, not run on hardware
//
#include <hip/hip_runtime.h>

#define B_TOT 2048
#define T_LEN 48
#define CH 35
#define HIDN 128
#define NUMQ 19
#define BROWS 16
#define THREADS 256   // 8 waves per block

typedef __attribute__((ext_vector_type(16))) __bf16 v16bf;
typedef __attribute__((ext_vector_type(8)))  float v8f;
typedef __attribute__((ext_vector_type(4)))  unsigned int v4u;

// fragment region offsets (ushort units); tile = nt*Ktiles + kt, 512 bf16 per tile
#define FR_DH   0        // Kt=2  Nt=8   ->  8192
#define FR_HR   8192     // Kt=4  Nt=3   ->  6144
#define FR_FR   14336    // Kt=2  Nt=3   ->  3072
#define FR_COMB 17408    // Kt=3  Nt=42  -> 64512
#define FR_IH   81920    // Kt=3  Nt=32  -> 49152
#define FR_HH   131072   // Kt=4  Nt=32  -> 65536
#define FR_TOT  196608

__device__ __forceinline__ unsigned short f2bf(float f) {
  union { float f; unsigned u; } x; x.f = f;
  unsigned r = x.u + 0x7fffu + ((x.u >> 16) & 1u);   // round-to-nearest-even
  return (unsigned short)(r >> 16);
}
__device__ __forceinline__ float sigmf(float x) { return 1.0f / (1.0f + __expf(-x)); }

// A fragment (16x32 bf16) from LDS, pair-packed row-major with pitch (bf16 units)
__device__ __forceinline__ v16bf load_a(const unsigned short* s, int pitch, int kt, int lane) {
  const unsigned short* p = s + (lane & 15) * pitch + kt * 32 + (lane >> 4) * 8;
  union { v16bf v; v4u q[2]; } u;
  u.q[0] = *(const v4u*)(p);
  u.q[1] = *(const v4u*)(p + 16);
  return u.v;
}
// B fragment (32x16 bf16) from pre-swizzled global storage
__device__ __forceinline__ v16bf load_b(const unsigned short* g, int tile, int lane) {
  const unsigned short* p = g + tile * 512 + lane * 16;
  union { v16bf v; v4u q[2]; } u;
  u.q[0] = *(const v4u*)(p);
  u.q[1] = *(const v4u*)(p + 8);
  return u.v;
}
__device__ __forceinline__ v8f wmma_bf(v16bf a, v16bf b, v8f c) {
  return __builtin_amdgcn_wmma_f32_16x16x32_bf16(false, a, false, b, (short)0, c, false, false);
}

// ---------------- prologue: swizzle weights to WMMA B-fragment layout ----------------
__global__ void pack_weights(const float* __restrict__ w_dh, const float* __restrict__ w_hr,
                             const float* __restrict__ w_fr, const float* __restrict__ w_comb,
                             const float* __restrict__ w_ih, const float* __restrict__ w_hh,
                             unsigned short* __restrict__ frag, float* __restrict__ zbuf) {
  if (blockIdx.x == 0 && threadIdx.x < 96) zbuf[threadIdx.x] = 0.0f;  // numer[48]+msum[48]
  int e = blockIdx.x * blockDim.x + threadIdx.x;
  if (e >= FR_TOT) return;
  int base, Ktiles, K, N, ldw; const float* W; bool zdiag = false;
  if (e < FR_HR)        { base = FR_DH;   Ktiles = 2; K = CH;    N = HIDN;    W = w_dh;   ldw = CH;   }
  else if (e < FR_FR)   { base = FR_HR;   Ktiles = 4; K = HIDN;  N = CH;      W = w_hr;   ldw = HIDN; }
  else if (e < FR_COMB) { base = FR_FR;   Ktiles = 2; K = CH;    N = CH;      W = w_fr;   ldw = CH;   zdiag = true; }
  else if (e < FR_IH)   { base = FR_COMB; Ktiles = 3; K = 2*CH;  N = NUMQ*CH; W = w_comb; ldw = 2*CH; }
  else if (e < FR_HH)   { base = FR_IH;   Ktiles = 3; K = 2*CH;  N = 4*HIDN;  W = w_ih;   ldw = 2*CH; }
  else                  { base = FR_HH;   Ktiles = 4; K = HIDN;  N = 4*HIDN;  W = w_hh;   ldw = HIDN; }
  int rel = e - base;
  int tile = rel >> 9, r = rel & 511;
  int lane = r >> 4, jj = r & 15;
  int nt = tile / Ktiles, kt = tile - nt * Ktiles;
  int n = nt * 16 + (lane & 15);
  int k = kt * 32 + (lane >> 4) * 8 + (jj & 7) + ((jj >> 3) << 4);
  float val = 0.0f;
  if (k < K && n < N) { val = W[n * ldw + k]; if (zdiag && k == n) val = 0.0f; }
  frag[e] = f2bf(val);
}

// ---------------- main recurrent kernel (8 waves / 16 batch rows) ----------------
__global__ __launch_bounds__(THREADS, 4)
__attribute__((amdgpu_waves_per_eu(4)))
void brits_main(
    const float* __restrict__ x, const float* __restrict__ mask, const float* __restrict__ delta,
    const float* __restrict__ b_dh, const float* __restrict__ w_dx, const float* __restrict__ b_dx,
    const float* __restrict__ b_hr, const float* __restrict__ b_fr,
    const float* __restrict__ b_comb, const float* __restrict__ b_ih, const float* __restrict__ b_hh,
    const unsigned short* __restrict__ frag,
    float* __restrict__ wsNumer, float* __restrict__ wsMsum,
    float* __restrict__ imp, float* __restrict__ ens)
{
  __shared__ float sXT[BROWS*36], sMT[BROWS*36], sGX[BROWS*36], sSUM[BROWS*36];
  __shared__ float sXH[BROWS*48], sZH[BROWS*48];
  __shared__ float sH32[BROWS*HIDN], sC[BROWS*HIDN];
  __shared__ float sBiasH[HIDN], sBiasHR[48], sBiasFR[48], sBiasDX[36], sDiag[36];
  __shared__ float sBiasComb[672], sBiasG[512];
  __shared__ float sRed[2];
  __shared__ __align__(16) unsigned short sHbf[BROWS*HIDN];  // bf16 copy of h
  __shared__ __align__(16) unsigned short sAh[BROWS*96];     // bf16 A-matrix staging

  const int tid  = threadIdx.x;
  const int lane = tid & 31;
  const int wave = tid >> 5;           // 0..7
  const int nl   = lane & 15;
  const int mh   = (lane >> 4) << 3;   // C/D fragment row offset
  const int b0   = blockIdx.x * BROWS;

  for (int i = tid; i < HIDN; i += THREADS) sBiasH[i] = b_dh[i];
  for (int i = tid; i < 48; i += THREADS) { sBiasHR[i] = i < CH ? b_hr[i] : 0.f; sBiasFR[i] = i < CH ? b_fr[i] : 0.f; }
  for (int i = tid; i < 36; i += THREADS) { sBiasDX[i] = i < CH ? b_dx[i] : 0.f; sDiag[i] = i < CH ? w_dx[i*CH+i] : 0.f; }
  for (int i = tid; i < 672; i += THREADS) sBiasComb[i] = i < NUMQ*CH ? b_comb[i] : 0.f;
  for (int i = tid; i < 512; i += THREADS) sBiasG[i] = b_ih[i] + b_hh[i];
  for (int i = tid; i < BROWS*HIDN; i += THREADS) { sH32[i] = 0.f; sC[i] = 0.f; sHbf[i] = 0; }
  __syncthreads();

  #pragma unroll 1
  for (int t = 0; t < T_LEN; ++t) {
    // ---- phase A: load inputs, gamma_x, stage dt (K padded to 64) ----
    for (int i = tid; i < BROWS*36; i += THREADS) sSUM[i] = 0.f;
    if (tid < 2) sRed[tid] = 0.f;
    for (int i = tid; i < BROWS*64; i += THREADS) {
      int m = i >> 6, k = i & 63;
      unsigned short bv = 0;
      if (k < CH) {
        size_t g = ((size_t)(b0 + m) * T_LEN + t) * CH + k;
        float xv = x[g], mv = mask[g], dv = delta[g];
        sXT[m*36+k] = xv; sMT[m*36+k] = mv;
        sGX[m*36+k] = __expf(-fmaxf(dv * sDiag[k] + sBiasDX[k], 0.f));
        bv = f2bf(dv);
      }
      sAh[m*96+k] = bv;
    }
    if (t + 1 < T_LEN && tid < BROWS) {   // prefetch next step's inputs
      size_t g = ((size_t)(b0 + tid) * T_LEN + (t + 1)) * CH;
      __builtin_prefetch(&x[g], 0, 1);
      __builtin_prefetch(&mask[g], 0, 1);
      __builtin_prefetch(&delta[g], 0, 1);
    }
    __syncthreads();

    // ---- gamma_h = exp(-relu(dt@W_dh^T+b)); h *= gamma_h   (8 N-tiles, 1/wave) ----
    {
      v16bf a0 = load_a(sAh, 96, 0, lane), a1 = load_a(sAh, 96, 1, lane);
      int nt = wave;
      float bv = sBiasH[nt*16 + nl];
      v8f acc;
      #pragma unroll
      for (int j = 0; j < 8; ++j) acc[j] = bv;
      acc = wmma_bf(a0, load_b(frag + FR_DH, nt*2 + 0, lane), acc);
      acc = wmma_bf(a1, load_b(frag + FR_DH, nt*2 + 1, lane), acc);
      int ncol = nt*16 + nl;
      #pragma unroll
      for (int j = 0; j < 8; ++j) {
        int m = j + mh;
        float gm = __expf(-fmaxf(acc[j], 0.f));
        float hn = sH32[m*HIDN + ncol] * gm;
        sH32[m*HIDN + ncol] = hn;
        sHbf[m*HIDN + ncol] = f2bf(hn);
      }
    }
    __syncthreads();

    // ---- x_h = h @ W_hr^T + b_hr  (3 N-tiles, waves 0..2) ----
    if (wave < 3) {
      v16bf ah[4];
      #pragma unroll
      for (int kt = 0; kt < 4; ++kt) ah[kt] = load_a(sHbf, HIDN, kt, lane);
      int nt = wave;
      float bv = sBiasHR[nt*16 + nl];
      v8f acc;
      #pragma unroll
      for (int j = 0; j < 8; ++j) acc[j] = bv;
      #pragma unroll
      for (int kt = 0; kt < 4; ++kt) acc = wmma_bf(ah[kt], load_b(frag + FR_HR, nt*4 + kt, lane), acc);
      int ncol = nt*16 + nl;
      #pragma unroll
      for (int j = 0; j < 8; ++j) sXH[(j + mh)*48 + ncol] = acc[j];
    }
    __syncthreads();

    // ---- stage x_c = m*x + (1-m)*x_h ----
    for (int i = tid; i < BROWS*64; i += THREADS) {
      int m = i >> 6, k = i & 63;
      unsigned short bv = 0;
      if (k < CH) {
        float mv = sMT[m*36+k];
        bv = f2bf(mv * sXT[m*36+k] + (1.f - mv) * sXH[m*48+k]);
      }
      sAh[m*96+k] = bv;
    }
    __syncthreads();

    // ---- z_h = x_c @ fr_w^T + b_fr  (3 N-tiles) ----
    if (wave < 3) {
      v16bf a0 = load_a(sAh, 96, 0, lane), a1 = load_a(sAh, 96, 1, lane);
      int nt = wave;
      float bv = sBiasFR[nt*16 + nl];
      v8f acc;
      #pragma unroll
      for (int j = 0; j < 8; ++j) acc[j] = bv;
      acc = wmma_bf(a0, load_b(frag + FR_FR, nt*2 + 0, lane), acc);
      acc = wmma_bf(a1, load_b(frag + FR_FR, nt*2 + 1, lane), acc);
      int ncol = nt*16 + nl;
      #pragma unroll
      for (int j = 0; j < 8; ++j) sZH[(j + mh)*48 + ncol] = acc[j];
    }
    __syncthreads();

    // ---- stage cat = [gamma_x, m] (K padded to 96) ----
    for (int i = tid; i < BROWS*96; i += THREADS) {
      int m = i / 96, k = i - m * 96;
      unsigned short bv = 0;
      if (k < CH) bv = f2bf(sGX[m*36+k]);
      else if (k < 2*CH) bv = f2bf(sMT[m*36 + (k - CH)]);
      sAh[m*96+k] = bv;
    }
    __syncthreads();

    // ---- alpha = cat @ W_comb^T + b; temp/ens/qloss; sum for mean ----
    float qAcc = 0.f;
    {
      v16bf ac0 = load_a(sAh, 96, 0, lane), ac1 = load_a(sAh, 96, 1, lane), ac2 = load_a(sAh, 96, 2, lane);
      #pragma unroll 1
      for (int nt = wave; nt < 42; nt += 8) {
        float bv = sBiasComb[nt*16 + nl];
        v8f acc;
        #pragma unroll
        for (int j = 0; j < 8; ++j) acc[j] = bv;
        acc = wmma_bf(ac0, load_b(frag + FR_COMB, nt*3 + 0, lane), acc);
        acc = wmma_bf(ac1, load_b(frag + FR_COMB, nt*3 + 1, lane), acc);
        acc = wmma_bf(ac2, load_b(frag + FR_COMB, nt*3 + 2, lane), acc);
        int o = nt*16 + nl;
        if (o < NUMQ*CH) {
          int nq = o / CH, c = o - nq * CH;
          float q = 0.05f * (float)(nq + 1);
          #pragma unroll
          for (int j = 0; j < 8; ++j) {
            int m = j + mh;
            float a  = acc[j];
            float zv = sZH[m*48+c], xh = sXH[m*48+c];
            float xv = sXT[m*36+c], mv = sMT[m*36+c];
            float temp = a * zv + (1.f - a) * xh;
            float ind = ((xv <= temp) ? 1.f : 0.f) - q;
            qAcc += fabsf((temp - xv) * mv * ind);
            float ev = mv * xv + (1.f - mv) * temp;
            ens[(((size_t)(b0 + m) * NUMQ + nq) * T_LEN + t) * CH + c] = ev;
            atomicAdd(&sSUM[m*36+c], temp);
          }
        }
      }
    }
    __syncthreads();

    // ---- elementwise: errors, msum, c_c, imputations; stage cat2=[c_c,m] ----
    {
      float eAcc = 0.f, mAcc = 0.f;
      for (int i = tid; i < BROWS*96; i += THREADS) {
        int m = i / 96, k = i - m * 96;
        unsigned short bv = 0;
        if (k < CH) {
          float xv = sXT[m*36+k], mv = sMT[m*36+k];
          float meanT = sSUM[m*36+k] * (1.f / NUMQ);
          float xh = sXH[m*48+k], zh = sZH[m*48+k];
          eAcc += (fabsf(xv - xh) + fabsf(xv - zh) + fabsf(xv - meanT)) * mv;
          mAcc += mv;
          float cc = mv * xv + (1.f - mv) * meanT;
          imp[((size_t)(b0 + m) * T_LEN + t) * CH + k] = cc;
          bv = f2bf(cc);
        } else if (k < 2*CH) bv = f2bf(sMT[m*36 + (k - CH)]);
        sAh[m*96+k] = bv;
      }
      atomicAdd(&sRed[0], eAcc + qAcc * (1.f / NUMQ));
      atomicAdd(&sRed[1], mAcc);
    }
    __syncthreads();
    if (tid == 0) {
      atomicAdd(&wsNumer[t], sRed[0]);
      atomicAdd(&wsMsum[t], sRed[1]);
    }

    // ---- LSTM gates: [c_c,m]@W_ih^T + h@W_hh^T + b ----
    // With 8 waves, wave w owns tiles {w, w+8, w+16, w+24} = i/f/g/o of columns [16w,16w+16):
    // one pass, 4 accumulators live.
    {
      v16bf ag0 = load_a(sAh, 96, 0, lane), ag1 = load_a(sAh, 96, 1, lane), ag2 = load_a(sAh, 96, 2, lane);
      v16bf ah0 = load_a(sHbf, HIDN, 0, lane), ah1 = load_a(sHbf, HIDN, 1, lane);
      v16bf ah2 = load_a(sHbf, HIDN, 2, lane), ah3 = load_a(sHbf, HIDN, 3, lane);
      __syncthreads();   // A fragments captured in registers before any h update
      v8f acc4[4];
      #pragma unroll
      for (int gg = 0; gg < 4; ++gg) {
        int nt = wave + 8 * gg;        // i, f, g, o tiles
        float bv = sBiasG[nt*16 + nl];
        v8f a;
        #pragma unroll
        for (int j = 0; j < 8; ++j) a[j] = bv;
        a = wmma_bf(ag0, load_b(frag + FR_IH, nt*3 + 0, lane), a);
        a = wmma_bf(ag1, load_b(frag + FR_IH, nt*3 + 1, lane), a);
        a = wmma_bf(ag2, load_b(frag + FR_IH, nt*3 + 2, lane), a);
        a = wmma_bf(ah0, load_b(frag + FR_HH, nt*4 + 0, lane), a);
        a = wmma_bf(ah1, load_b(frag + FR_HH, nt*4 + 1, lane), a);
        a = wmma_bf(ah2, load_b(frag + FR_HH, nt*4 + 2, lane), a);
        a = wmma_bf(ah3, load_b(frag + FR_HH, nt*4 + 3, lane), a);
        acc4[gg] = a;
      }
      int col = wave * 16 + nl;        // column 0..127
      #pragma unroll
      for (int j = 0; j < 8; ++j) {
        int m = j + mh;
        float ig = sigmf(acc4[0][j]);
        float fg = sigmf(acc4[1][j]);
        float gv = tanhf(acc4[2][j]);
        float og = sigmf(acc4[3][j]);
        float cn = fg * sC[m*HIDN + col] + ig * gv;
        float hn = og * tanhf(cn);
        sC[m*HIDN + col]  = cn;
        sH32[m*HIDN + col] = hn;
        sHbf[m*HIDN + col] = f2bf(hn);
      }
    }
    __syncthreads();
  }
}

// ---------------- finalize: loss = mean_t(numer/(msum+1e-5)) ----------------
__global__ void finalize_loss(const float* __restrict__ numer, const float* __restrict__ msum,
                              float* __restrict__ out) {
  __shared__ float s[T_LEN];
  int t = threadIdx.x;
  if (t < T_LEN) s[t] = numer[t] / (msum[t] + 1e-5f);
  __syncthreads();
  if (t == 0) {
    float a = 0.f;
    for (int i = 0; i < T_LEN; ++i) a += s[i];
    out[0] = a / (float)T_LEN;
  }
}

extern "C" void kernel_launch(void* const* d_in, const int* in_sizes, int n_in,
                              void* d_out, int out_size, void* d_ws, size_t ws_size,
                              hipStream_t stream) {
  (void)in_sizes; (void)n_in; (void)out_size; (void)ws_size;
  const float* x      = (const float*)d_in[0];
  const float* maskp  = (const float*)d_in[1];
  const float* delta  = (const float*)d_in[2];
  const float* w_dh   = (const float*)d_in[3];
  const float* b_dh   = (const float*)d_in[4];
  const float* w_dx   = (const float*)d_in[5];
  const float* b_dx   = (const float*)d_in[6];
  const float* w_hr   = (const float*)d_in[7];
  const float* b_hr   = (const float*)d_in[8];
  const float* w_fr   = (const float*)d_in[9];
  const float* b_fr   = (const float*)d_in[10];
  const float* w_comb = (const float*)d_in[11];
  const float* b_comb = (const float*)d_in[12];
  const float* w_ih   = (const float*)d_in[13];
  const float* w_hh   = (const float*)d_in[14];
  const float* b_ih   = (const float*)d_in[15];
  const float* b_hh   = (const float*)d_in[16];

  float* wsNumer = (float*)d_ws;
  float* wsMsum  = wsNumer + T_LEN;
  unsigned short* frag = (unsigned short*)((char*)d_ws + 512);

  float* out = (float*)d_out;
  float* imp = out + 1;
  float* ens = imp + (size_t)B_TOT * T_LEN * CH;

  pack_weights<<<dim3((FR_TOT + 255) / 256), dim3(256), 0, stream>>>(
      w_dh, w_hr, w_fr, w_comb, w_ih, w_hh, frag, wsNumer);
  brits_main<<<dim3(B_TOT / BROWS), dim3(THREADS), 0, stream>>>(
      x, maskp, delta, b_dh, w_dx, b_dx, b_hr, b_fr, b_comb, b_ih, b_hh,
      frag, wsNumer, wsMsum, imp, ens);
  finalize_loss<<<dim3(1), dim3(64), 0, stream>>>(wsNumer, wsMsum, out);
}
